// BMM_3204045603931
// MI455X (gfx1250) — matrix operand order repaired and checked by reading, not compiled, not run on hardware
//
#include <hip/hip_runtime.h>

#define NEXP   8
#define MROWS  1024   // rows per expert
#define DK     2048   // K dim; also N of act and of out
#define BM     128
#define BN     128
#define BK     32     // one 16x16x32 WMMA K-step per tile
#define ROWB   40     // LDS row stride in bf16 elements (80 B: 64 B data + 16 B pad)

typedef __attribute__((ext_vector_type(4)))  float  v4f;
typedef __attribute__((ext_vector_type(8)))  float  v8f;
typedef __attribute__((ext_vector_type(4)))  __bf16 v4bf;
typedef __attribute__((ext_vector_type(8)))  __bf16 v8bf;
typedef __attribute__((ext_vector_type(16))) __bf16 v16bf;

union Frag {
  v16bf v;
  v8bf  h[2];
};

template <bool GATEUP>
__global__ __launch_bounds__(256) void mlp_gemm(const float* __restrict__ Abase,
                                                const float* __restrict__ Bbase,
                                                float* __restrict__ Cbase,
                                                int nbstride) {
  // A: [m][k] packed bf16, hi/lo planes.  B: transposed [n][k] bf16, hi/lo planes.
  __shared__ __bf16 Ah[BM * ROWB], Al[BM * ROWB];
  __shared__ __bf16 Bh0[BN * ROWB], Bl0[BN * ROWB];
  __shared__ __bf16 Bh1[GATEUP ? (BN * ROWB) : 1], Bl1[GATEUP ? (BN * ROWB) : 1];

  const int tid  = threadIdx.x;
  const int lane = tid & 31;
  const int wid  = tid >> 5;  // 0..7
  const int e    = blockIdx.z;
  const int m0   = blockIdx.y * BM;
  const int n0   = blockIdx.x * BN;

  const float* A = Abase + (size_t)e * MROWS * DK + (size_t)m0 * DK;
  const float* B = Bbase + (size_t)e * DK * nbstride + n0;
  float*       C = Cbase + (size_t)e * MROWS * DK + (size_t)m0 * DK + n0;

  const int wm   = (wid & 3) * 32;  // wave M offset in block tile
  const int wn   = (wid >> 2) * 64; // wave N offset in block tile
  const int l16  = lane & 15;
  const int hsel = lane >> 4;       // 0: low half, 1: high half (ISA 16-bit layouts)

  v8f accg[2][4];
  v8f accu[2][4];
#pragma unroll
  for (int i = 0; i < 2; ++i)
#pragma unroll
    for (int j = 0; j < 4; ++j) {
      accg[i][j] = (v8f)0.0f;
      if constexpr (GATEUP) accu[i][j] = (v8f)0.0f;
    }

  // staging coordinates (float4 granularity)
  const int a_c = (tid & 7) * 4;   // A col within BK
  const int a_m = tid >> 3;        // A row, 32/round, 4 rounds
  const int b_n = (tid & 31) * 4;  // B col group (4 n values)
  const int b_k = (tid >> 5) * 4;  // B k group (4 k values)

  v4f pa[4], pg[4], pu[4];

  auto prefetch = [&](int k0) {
#pragma unroll
    for (int i = 0; i < 4; ++i)
      pa[i] = *(const v4f*)(A + (size_t)(a_m + 32 * i) * DK + k0 + a_c);
#pragma unroll
    for (int j = 0; j < 4; ++j) {
      const float* br = B + (size_t)(k0 + b_k + j) * nbstride;
      pg[j] = *(const v4f*)(br + b_n);
      if constexpr (GATEUP) pu[j] = *(const v4f*)(br + DK + b_n);
    }
  };

  prefetch(0);

  for (int k0 = 0; k0 < DK; k0 += BK) {
    if (k0) __syncthreads();  // previous tile's compute done -> LDS reusable

    // ---- stage A (hi/lo split), [m][k] packed ----
#pragma unroll
    for (int i = 0; i < 4; ++i) {
      v4bf h, l;
#pragma unroll
      for (int j = 0; j < 4; ++j) {
        const float  x  = pa[i][j];
        const __bf16 hh = (__bf16)x;
        h[j] = hh;
        l[j] = (__bf16)(x - (float)hh);
      }
      const int off = (a_m + 32 * i) * ROWB + a_c;  // 8B aligned
      *(v4bf*)&Ah[off] = h;
      *(v4bf*)&Al[off] = l;
    }
    // ---- stage B (hi/lo split), register-transposed to [n][k] ----
#pragma unroll
    for (int i = 0; i < 4; ++i) {  // over n
      v4bf hg, lg, uh, ul;
#pragma unroll
      for (int j = 0; j < 4; ++j) {  // over k
        const float  xg = pg[j][i];
        const __bf16 hh = (__bf16)xg;
        hg[j] = hh;
        lg[j] = (__bf16)(xg - (float)hh);
        if constexpr (GATEUP) {
          const float  xu = pu[j][i];
          const __bf16 uhh = (__bf16)xu;
          uh[j] = uhh;
          ul[j] = (__bf16)(xu - (float)uhh);
        }
      }
      const int off = (b_n + i) * ROWB + b_k;  // 8B aligned
      *(v4bf*)&Bh0[off] = hg;
      *(v4bf*)&Bl0[off] = lg;
      if constexpr (GATEUP) {
        *(v4bf*)&Bh1[off] = uh;
        *(v4bf*)&Bl1[off] = ul;
      }
    }
    __syncthreads();

    if (k0 + BK < DK) prefetch(k0 + BK);  // overlap next global loads with WMMAs

    // ---- A fragments: per lane two contiguous 16B chunks ----
    Frag ah[2], al[2];
#pragma unroll
    for (int mt = 0; mt < 2; ++mt) {
      const int off = (wm + mt * 16 + l16) * ROWB + hsel * 8;  // elements
      ah[mt].h[0] = *(const v8bf*)&Ah[off];
      ah[mt].h[1] = *(const v8bf*)&Ah[off + 16];
      al[mt].h[0] = *(const v8bf*)&Al[off];
      al[mt].h[1] = *(const v8bf*)&Al[off + 16];
    }
    // ---- per N-tile: B fragments + bf16x3 WMMA ----
#pragma unroll
    for (int nt = 0; nt < 4; ++nt) {
      const int off = (wn + nt * 16 + l16) * ROWB + hsel * 8;  // elements
      Frag bh, bl;
      bh.h[0] = *(const v8bf*)&Bh0[off];
      bh.h[1] = *(const v8bf*)&Bh0[off + 16];
      bl.h[0] = *(const v8bf*)&Bl0[off];
      bl.h[1] = *(const v8bf*)&Bl0[off + 16];
#pragma unroll
      for (int mt = 0; mt < 2; ++mt) {
        accg[mt][nt] = __builtin_amdgcn_wmma_f32_16x16x32_bf16(
            false, al[mt].v, false, bh.v, (short)0, accg[mt][nt], false, false);
        accg[mt][nt] = __builtin_amdgcn_wmma_f32_16x16x32_bf16(
            false, ah[mt].v, false, bl.v, (short)0, accg[mt][nt], false, false);
        accg[mt][nt] = __builtin_amdgcn_wmma_f32_16x16x32_bf16(
            false, ah[mt].v, false, bh.v, (short)0, accg[mt][nt], false, false);
      }
      if constexpr (GATEUP) {
        Frag uhf, ulf;
        uhf.h[0] = *(const v8bf*)&Bh1[off];
        uhf.h[1] = *(const v8bf*)&Bh1[off + 16];
        ulf.h[0] = *(const v8bf*)&Bl1[off];
        ulf.h[1] = *(const v8bf*)&Bl1[off + 16];
#pragma unroll
        for (int mt = 0; mt < 2; ++mt) {
          accu[mt][nt] = __builtin_amdgcn_wmma_f32_16x16x32_bf16(
              false, al[mt].v, false, uhf.v, (short)0, accu[mt][nt], false, false);
          accu[mt][nt] = __builtin_amdgcn_wmma_f32_16x16x32_bf16(
              false, ah[mt].v, false, ulf.v, (short)0, accu[mt][nt], false, false);
          accu[mt][nt] = __builtin_amdgcn_wmma_f32_16x16x32_bf16(
              false, ah[mt].v, false, uhf.v, (short)0, accu[mt][nt], false, false);
        }
      }
    }
  }

  // ---- epilogue: C/D layout -> VGPR r holds M=r (lanes 0-15) / r+8 (lanes 16-31)
  const int rsel = hsel << 3;
#pragma unroll
  for (int mt = 0; mt < 2; ++mt) {
#pragma unroll
    for (int nt = 0; nt < 4; ++nt) {
      v8f v = accg[mt][nt];
      if constexpr (GATEUP) {
        const v8f u = accu[mt][nt];
#pragma unroll
        for (int r = 0; r < 8; ++r) v[r] = u[r] * fmaxf(v[r], 0.0f);
      }
      const int col   = wn + nt * 16 + l16;
      const int rbase = wm + mt * 16 + rsel;
#pragma unroll
      for (int r = 0; r < 8; ++r) C[(size_t)(rbase + r) * DK + col] = v[r];
    }
  }
}

extern "C" void kernel_launch(void* const* d_in, const int* in_sizes, int n_in,
                              void* d_out, int out_size, void* d_ws, size_t ws_size,
                              hipStream_t stream) {
  const float* hidden = (const float*)d_in[0];  // (8192, 2048)
  const float* w1     = (const float*)d_in[1];  // (8, 2048, 4096)
  const float* w2     = (const float*)d_in[2];  // (8, 2048, 2048)
  float*       out    = (float*)d_out;          // (8192, 2048)
  float*       act    = (float*)d_ws;           // (8, 1024, 2048) fp32 = 64 MB scratch

  dim3 block(256);
  dim3 grid(DK / BN, MROWS / BM, NEXP);  // (16, 8, 8)

  // Stage 1: act = up * relu(gate); both halves of h @ W1 share the A tile.
  mlp_gemm<true><<<grid, block, 0, stream>>>(hidden, w1, act, 2 * DK);
  // Stage 2: out = act @ W2.
  mlp_gemm<false><<<grid, block, 0, stream>>>(act, w2, out, DK);
}
